// HTM_9904194585244
// MI455X (gfx1250) — matrix operand order repaired and checked by reading, not compiled, not run on hardware
//
#include <hip/hip_runtime.h>
#include <math.h>

// ---------------------------------------------------------------------------
// Problem constants (from reference)
// ---------------------------------------------------------------------------
#define BB      4
#define LL      2048
#define DMODEL  1024
#define DSTATE  16
#define DCONV   4
#define EINNER  2048                   // D_INNER
#define PROJ    (2 * EINNER)           // 4096
#define DTRANK  64
#define XDBLW   (DTRANK + 2 * DSTATE)  // 96
#define XDBLPAD 128                    // x_proj rows padded for unguarded DMA
#define NTOT    (BB * LL)              // 8192 token positions

typedef __attribute__((ext_vector_type(16))) __bf16 v16bf;
typedef __attribute__((ext_vector_type(8)))  float  v8f;

__device__ __forceinline__ unsigned short f32_to_bf16(float f) {
    unsigned int u = __float_as_uint(f);
    u += 0x7FFFu + ((u >> 16) & 1u);     // round to nearest even
    return (unsigned short)(u >> 16);
}
__device__ __forceinline__ float bf16_to_f32(unsigned short h) {
    return __uint_as_float((unsigned int)h << 16);
}

// CDNA5 async global->LDS DMA (ASYNCcnt-tracked), inline asm per ISA
// CDNA5 ISA (portable across ROCm7.2 / amdgpu-toolchain builtins).
__device__ __forceinline__ void async_ld_b128(unsigned lds_off, const void* g) {
    asm volatile("global_load_async_to_lds_b128 %0, %1, off"
                 :: "v"(lds_off), "v"(g) : "memory");
}
__device__ __forceinline__ void wait_async0() {
    asm volatile("s_wait_asynccnt 0x0" ::: "memory");
}

union Frag { uint4 q[2]; v16bf v; };

// ---------------------------------------------------------------------------
// Generic "NT" GEMM on bf16 operands:  C[n,m] = sum_k P[n,k] * Q[m,k]
//   P: Ntot x K row-major bf16;  Q: (padded-M) x K row-major bf16.
// Block: 256 threads = 8 wave32 waves; tile 64(n) x 64(m); K-step 32.
// K-tiles staged with global_load_async_to_lds_b128, double-buffered in LDS;
// fragments gathered with ds_load_b128 per the gfx1250 WMMA lane layouts;
// two v_wmma_f32_16x16x32_bf16 per wave per K-step (fp32 accumulate).
// MODE 0: plain (optional fp32 outF and/or bf16 outB)
// MODE 1: split at splitH: x-half -> bf16 outB, z-half -> fp32 outF
// MODE 2: softplus(acc + bias[m]) -> fp32 outF
// ---------------------------------------------------------------------------
template <int MODE>
__global__ __launch_bounds__(256) void gemm_nt_bf16(
    const unsigned short* __restrict__ P, int ldP,
    const unsigned short* __restrict__ Q, int ldQ,
    int Mtot, int K,
    float* __restrict__ outF, unsigned short* __restrict__ outB, int ldOut,
    int splitH, const float* __restrict__ bias)
{
    // 40-element (80 B) row stride keeps every b128 LDS access 16B-aligned.
    __shared__ __align__(16) unsigned short Plds[2][64 * 40];
    __shared__ __align__(16) unsigned short Qlds[2][64 * 40];

    const int tid  = threadIdx.x;
    const int lane = tid & 31;
    const int wave = tid >> 5;
    const int wr   = wave & 3;      // n sub-tile
    const int wc   = wave >> 2;     // m sub-tile (32 wide)
    const int hsel = lane >> 4;     // wave half -> K-slice select
    const int l15  = lane & 15;

    const int n0 = blockIdx.x * 64;
    const int m0 = blockIdx.y * 64;

    const int srow = tid >> 2;        // staged row 0..63
    const int scol = (tid & 3) * 8;   // bf16 col group: 8 elems = 16 B

    v8f acc0 = {};
    v8f acc1 = {};

    // one async b128 per thread per operand per K-tile (wave moves 512 B/instr)
    auto issue = [&](int k0, int buf) {
        async_ld_b128((unsigned)(size_t)&Plds[buf][srow * 40 + scol],
                      P + (size_t)(n0 + srow) * ldP + k0 + scol);
        async_ld_b128((unsigned)(size_t)&Qlds[buf][srow * 40 + scol],
                      Q + (size_t)(m0 + srow) * ldQ + k0 + scol);
    };

    issue(0, 0);
    int cur = 0;

    for (int k0 = 0; k0 < K; k0 += 32) {
        wait_async0();        // buf[cur] DMA complete (this wave)
        __syncthreads();      // visible to all waves; prior reads of buf[cur^1] done
        if (k0 + 32 < K)
            issue(k0 + 32, cur ^ 1);   // overlap next tile's DMA with compute

        // A (16x32): lanes 0-15 hold K 0..15; lanes 16-31 K 16..31
        Frag a, b0, b1;
        const unsigned short* pr = &Plds[cur][(wr * 16 + l15) * 40];
        a.q[0] = *(const uint4*)(pr + hsel * 16);
        a.q[1] = *(const uint4*)(pr + hsel * 16 + 8);
        // B (32x16): lanes 0-15 hold K 0..15; lanes 16-31 K 16..31 (N = lane&15)
        const unsigned short* q0 = &Qlds[cur][(wc * 32 + l15) * 40];
        b0.q[0] = *(const uint4*)(q0 + hsel * 16);
        b0.q[1] = *(const uint4*)(q0 + hsel * 16 + 8);
        const unsigned short* q1 = q0 + 16 * 40;
        b1.q[0] = *(const uint4*)(q1 + hsel * 16);
        b1.q[1] = *(const uint4*)(q1 + hsel * 16 + 8);

        acc0 = __builtin_amdgcn_wmma_f32_16x16x32_bf16(
                   false, a.v, false, b0.v, (short)0, acc0, false, false);
        acc1 = __builtin_amdgcn_wmma_f32_16x16x32_bf16(
                   false, a.v, false, b1.v, (short)0, acc1, false, false);
        cur ^= 1;
    }

    // ---- epilogue: D layout lane(l15)=N col, VGPR r => M row r + 8*hsel ----
    union { v8f v; float f[8]; } c0, c1;
    c0.v = acc0;
    c1.v = acc1;
    const int gm0 = m0 + wc * 32 + l15;
    const int gm1 = gm0 + 16;
    #pragma unroll
    for (int r = 0; r < 8; ++r) {
        const int gn = n0 + wr * 16 + hsel * 8 + r;
        float v0 = c0.f[r], v1 = c1.f[r];
        if (MODE == 0) {
            if (gm0 < Mtot) {
                if (outF) outF[(size_t)gn * ldOut + gm0] = v0;
                if (outB) outB[(size_t)gn * ldOut + gm0] = f32_to_bf16(v0);
            }
            if (gm1 < Mtot) {
                if (outF) outF[(size_t)gn * ldOut + gm1] = v1;
                if (outB) outB[(size_t)gn * ldOut + gm1] = f32_to_bf16(v1);
            }
        } else if (MODE == 1) {  // x-half -> bf16 (conv input), z-half -> fp32
            if (gm0 < splitH) outB[(size_t)gn * splitH + gm0] = f32_to_bf16(v0);
            else              outF[(size_t)gn * splitH + gm0 - splitH] = v0;
            if (gm1 < splitH) outB[(size_t)gn * splitH + gm1] = f32_to_bf16(v1);
            else              outF[(size_t)gn * splitH + gm1 - splitH] = v1;
        } else {                 // MODE 2: softplus(acc + bias[m])
            if (gm0 < Mtot) {
                float s = v0 + bias[gm0];
                outF[(size_t)gn * ldOut + gm0] = (s > 20.f) ? s : log1pf(__expf(s));
            }
            if (gm1 < Mtot) {
                float s = v1 + bias[gm1];
                outF[(size_t)gn * ldOut + gm1] = (s > 20.f) ? s : log1pf(__expf(s));
            }
        }
    }
}

// ---------------------------------------------------------------------------
// fp32 -> bf16 weight/activation conversion
// ---------------------------------------------------------------------------
__global__ __launch_bounds__(256) void cvt_f32_bf16_kernel(
    const float* __restrict__ src, unsigned short* __restrict__ dst, int n)
{
    const int i = blockIdx.x * 256 + threadIdx.x;
    if (i < n) dst[i] = f32_to_bf16(src[i]);
}

// x_proj (96 x E) -> zero-padded (128 x E) bf16 so GEMM DMA needs no guards
__global__ __launch_bounds__(256) void cvt_xproj_kernel(
    const float* __restrict__ src, unsigned short* __restrict__ dst)
{
    const int i = blockIdx.x * 256 + threadIdx.x;   // over 128*E
    const int r = i >> 11;                          // E == 2048
    const int c = i & (EINNER - 1);
    dst[i] = (r < XDBLW) ? f32_to_bf16(src[(size_t)r * EINNER + c]) : (unsigned short)0;
}

// ---------------------------------------------------------------------------
// Depthwise causal conv (width 4) + bias + SiLU; bf16 in, fp32 + bf16 out.
// (B,L,E) layout: lanes hit adjacent channels, all taps coalesced.
// ---------------------------------------------------------------------------
__global__ __launch_bounds__(256) void conv_silu_kernel(
    const unsigned short* __restrict__ xin,  // (B,L,E) bf16
    const float* __restrict__ cw,            // (E,4)
    const float* __restrict__ cb,            // (E)
    float* __restrict__ xc,                  // (B,L,E) fp32 (scan)
    unsigned short* __restrict__ xcb)        // (B,L,E) bf16 (x_proj GEMM)
{
    const int bl = blockIdx.x;               // b*L + l
    const int b  = bl / LL;
    const int l  = bl % LL;
    #pragma unroll
    for (int i = 0; i < EINNER / 256; ++i) {
        const int e = threadIdx.x + i * 256;
        float acc = cb[e];
        const float4 w = *(const float4*)&cw[e * 4];
        const float wv[4] = {w.x, w.y, w.z, w.w};
        #pragma unroll
        for (int t = 0; t < DCONV; ++t) {
            const int ls = l - (DCONV - 1) + t;
            if (ls >= 0)
                acc += bf16_to_f32(xin[((size_t)b * LL + ls) * EINNER + e]) * wv[t];
        }
        const float s = acc / (1.f + __expf(-acc));      // SiLU
        const size_t o = ((size_t)b * LL + l) * EINNER + e;
        xc[o]  = s;
        xcb[o] = f32_to_bf16(s);
    }
}

// ---------------------------------------------------------------------------
// Selective scan: one thread per channel (b,e); h[16] in VGPRs; per-step
// B_t/C_t broadcast via LDS. Fuses D-skip + SiLU(z); writes bf16 directly
// into the concatenated buffer consumed by the agg GEMM.
// ---------------------------------------------------------------------------
__global__ __launch_bounds__(256) void scan_kernel(
    const float* __restrict__ delta,  // (B,L,E) post-softplus
    const float* __restrict__ xc,     // (B,L,E) post conv+silu
    const float* __restrict__ xdbl,   // (B,L,96): [64:80)=B [80:96)=C
    const float* __restrict__ zb,     // (B,L,E)
    const float* __restrict__ A_log,  // (E,16)
    const float* __restrict__ Dp,     // (E)
    unsigned short* __restrict__ ycat,// (B,L,2E) bf16
    int kmod)
{
    __shared__ float BC[32];
    const int b = blockIdx.y;
    const int e = blockIdx.x * 256 + threadIdx.x;

    float A[DSTATE], h[DSTATE];
    #pragma unroll
    for (int n = 0; n < DSTATE; ++n) {
        A[n] = -__expf(A_log[(size_t)e * DSTATE + n]);
        h[n] = 0.f;
    }
    const float Dv = Dp[e];

    for (int l = 0; l < LL; ++l) {
        const size_t row = (size_t)b * LL + l;
        if (threadIdx.x < 32)
            BC[threadIdx.x] = xdbl[row * XDBLW + DTRANK + threadIdx.x];
        __syncthreads();

        const float dlt = delta[row * EINNER + e];
        const float xv  = xc  [row * EINNER + e];
        const float zv  = zb  [row * EINNER + e];
        const float dx  = dlt * xv;
        float y = 0.f;
        #pragma unroll
        for (int n = 0; n < DSTATE; ++n) {
            h[n] = h[n] * __expf(dlt * A[n]) + dx * BC[n];
            y   += h[n] * BC[DSTATE + n];
        }
        const float sz = zv / (1.f + __expf(-zv));       // SiLU(z)
        ycat[row * (2 * EINNER) + (size_t)kmod * EINNER + e] =
            f32_to_bf16((y + xv * Dv) * sz);
        __syncthreads();
    }
}

// ---------------------------------------------------------------------------
// Host-side orchestration (all on `stream`, graph-capture safe).
// ---------------------------------------------------------------------------
extern "C" void kernel_launch(void* const* d_in, const int* in_sizes, int n_in,
                              void* d_out, int out_size, void* d_ws, size_t ws_size,
                              hipStream_t stream) {
    (void)in_sizes; (void)n_in; (void)out_size; (void)ws_size;

    const float* hs        = (const float*)d_in[0];
    const float* in_proj_w = (const float*)d_in[1];
    const float* conv_w    = (const float*)d_in[2];
    const float* conv_b    = (const float*)d_in[3];
    const float* x_proj_w  = (const float*)d_in[4];
    const float* dt_w      = (const float*)d_in[5];
    const float* dt_b      = (const float*)d_in[6];
    const float* A_log     = (const float*)d_in[7];
    const float* D_param   = (const float*)d_in[8];
    const float* agg_w     = (const float*)d_in[9];
    const float* out_w     = (const float*)d_in[10];
    float*       out       = (float*)d_out;

    char* ws = (char*)d_ws;
    size_t off = 0;
    auto alloc = [&](size_t bytes) {
        void* p = ws + off;
        off += (bytes + 255) & ~(size_t)255;
        return p;
    };
    // bf16 weight copies
    unsigned short* hsb  = (unsigned short*)alloc((size_t)NTOT * DMODEL * 2);
    unsigned short* ipwb = (unsigned short*)alloc((size_t)PROJ * DMODEL * 2);
    unsigned short* xpwb = (unsigned short*)alloc((size_t)XDBLPAD * EINNER * 2);
    unsigned short* dtwb = (unsigned short*)alloc((size_t)2 * EINNER * DTRANK * 2);
    unsigned short* agwb = (unsigned short*)alloc((size_t)EINNER * PROJ * 2);
    unsigned short* otwb = (unsigned short*)alloc((size_t)DMODEL * EINNER * 2);
    // activations
    unsigned short* xbuf  = (unsigned short*)alloc((size_t)NTOT * EINNER * 2); // x half, bf16
    float*          zbuf  = (float*)alloc((size_t)NTOT * EINNER * 4);          // z half
    float*          xcbuf = (float*)alloc((size_t)NTOT * EINNER * 4);
    unsigned short* xcb   = (unsigned short*)alloc((size_t)NTOT * EINNER * 2);
    float*          dlt   = (float*)alloc((size_t)NTOT * EINNER * 4);
    float*          xdbl  = (float*)alloc((size_t)NTOT * XDBLW * 4);
    unsigned short* xdblb = (unsigned short*)alloc((size_t)NTOT * XDBLW * 2);
    unsigned short* ycat  = (unsigned short*)alloc((size_t)NTOT * PROJ * 2);
    unsigned short* aggb  = xbuf;   // xbuf is dead after both conv passes

    const dim3 blk(256);
    auto cvt = [&](const float* s, unsigned short* d, size_t n) {
        cvt_f32_bf16_kernel<<<dim3((unsigned)((n + 255) / 256)), blk, 0, stream>>>(s, d, (int)n);
    };

    // 0) one-shot fp32->bf16 conversions (weights + hidden states)
    cvt(hs,        hsb,  (size_t)NTOT * DMODEL);
    cvt(in_proj_w, ipwb, (size_t)PROJ * DMODEL);
    cvt_xproj_kernel<<<dim3((XDBLPAD * EINNER) / 256), blk, 0, stream>>>(x_proj_w, xpwb);
    cvt(dt_w,      dtwb, (size_t)2 * EINNER * DTRANK);
    cvt(agg_w,     agwb, (size_t)EINNER * PROJ);
    cvt(out_w,     otwb, (size_t)DMODEL * EINNER);

    // 1) in_proj: xz = in_proj_w @ hs^T ; x half -> bf16, z half -> fp32
    gemm_nt_bf16<1><<<dim3(NTOT / 64, PROJ / 64), blk, 0, stream>>>(
        hsb, DMODEL, ipwb, DMODEL, PROJ, DMODEL,
        zbuf, xbuf, EINNER, EINNER, nullptr);

    for (int k = 0; k < 2; ++k) {
        // 2) depthwise causal conv + SiLU
        conv_silu_kernel<<<dim3(NTOT), blk, 0, stream>>>(
            xbuf, conv_w + (size_t)k * EINNER * DCONV,
            conv_b + (size_t)k * EINNER, xcbuf, xcb);

        // 3) x_dbl = x_proj_w @ x  (M=96; Q padded to 128 rows, stores guarded)
        gemm_nt_bf16<0><<<dim3(NTOT / 64, 2), blk, 0, stream>>>(
            xcb, EINNER, xpwb, EINNER, XDBLW, EINNER,
            xdbl, xdblb, XDBLW, 0, nullptr);

        // 4) delta = softplus(dt_w @ dt_low + dt_b)   (K = 64)
        gemm_nt_bf16<2><<<dim3(NTOT / 64, EINNER / 64), blk, 0, stream>>>(
            xdblb, XDBLW, dtwb + (size_t)k * EINNER * DTRANK, DTRANK,
            EINNER, DTRANK, dlt, nullptr, EINNER, 0,
            dt_b + (size_t)k * EINNER);

        // 5) selective scan + gating -> concatenated bf16 branch output
        scan_kernel<<<dim3(EINNER / 256, BB), blk, 0, stream>>>(
            dlt, xcbuf, xdbl, zbuf,
            A_log + (size_t)k * EINNER * DSTATE,
            D_param + (size_t)k * EINNER, ycat, k);
    }

    // 6) aggregated = ycat @ agg_w^T  (bf16 out, feeds final GEMM)
    gemm_nt_bf16<0><<<dim3(NTOT / 64, EINNER / 64), blk, 0, stream>>>(
        ycat, PROJ, agwb, PROJ, EINNER, PROJ,
        nullptr, aggb, EINNER, 0, nullptr);

    // 7) out = aggregated @ out_w^T  (fp32 result)
    gemm_nt_bf16<0><<<dim3(NTOT / 64, DMODEL / 64), blk, 0, stream>>>(
        aggb, EINNER, otwb, EINNER, DMODEL, EINNER,
        out, nullptr, DMODEL, 0, nullptr);
}
